// Model_38001870635195
// MI455X (gfx1250) — matrix operand order repaired and checked by reading, not compiled, not run on hardware
//
#include <hip/hip_runtime.h>
#include <math.h>

// ---------------------------------------------------------------------------
// MI455X (gfx1250, wave32) implementation.
// Heavy ops (conv-as-GEMM, sim graph, graph aggregation) -> v_wmma_f32_16x16x32_bf16
// with 4xN register blocking (one wave = 16x64 output strip, A-fragment reuse x4).
// Everything else -> wave32 VALU kernels with shuffle reductions.
// ---------------------------------------------------------------------------

typedef __attribute__((ext_vector_type(16))) __bf16 v16bf;
typedef __attribute__((ext_vector_type(8)))  __bf16 v8bf;
typedef __attribute__((ext_vector_type(8)))  float  v8f;

#define EPSF 1e-12f
#define TEMP 0.07f

// ------------------------------ wave32 helpers ------------------------------
__device__ __forceinline__ float wave_sum(float v) {
#pragma unroll
  for (int m = 16; m >= 1; m >>= 1) v += __shfl_xor(v, m, 32);
  return v;
}
__device__ __forceinline__ int wave_sum_i(int v) {
#pragma unroll
  for (int m = 16; m >= 1; m >>= 1) v += __shfl_xor(v, m, 32);
  return v;
}
__device__ __forceinline__ float wave_max(float v) {
#pragma unroll
  for (int m = 16; m >= 1; m >>= 1) v = fmaxf(v, __shfl_xor(v, m, 32));
  return v;
}
__device__ __forceinline__ float wave_min(float v) {
#pragma unroll
  for (int m = 16; m >= 1; m >>= 1) v = fminf(v, __shfl_xor(v, m, 32));
  return v;
}
// order-preserving float -> uint key (total order, for k-th-largest search)
__device__ __forceinline__ unsigned sortkey(float f) {
  unsigned u = __float_as_uint(f);
  return (u & 0x80000000u) ? ~u : (u | 0x80000000u);
}
// concat two v8bf chunks into a v16bf A fragment (register-only)
__device__ __forceinline__ v16bf cat16(v8bf a0, v8bf a1) {
  return __builtin_shufflevector(a0, a1, 0, 1, 2, 3, 4, 5, 6, 7,
                                         8, 9, 10, 11, 12, 13, 14, 15);
}

// ---------------------------------------------------------------------------
// Kernel 0: repack conv weights fp32 [H,D,3] -> bf16 [3,H,D] (contiguous d)
// ---------------------------------------------------------------------------
__global__ void repack_w(const float* __restrict__ w, __bf16* __restrict__ wpk,
                         int H, int D) {
  const int total = 3 * H * D;
  for (int idx = blockIdx.x * blockDim.x + threadIdx.x; idx < total;
       idx += gridDim.x * blockDim.x) {
    const int d  = idx % D;
    const int h  = (idx / D) % H;
    const int kt = idx / (D * H);
    wpk[idx] = (__bf16)w[((size_t)h * D + d) * 3 + kt];
  }
}

// ---------------------------------------------------------------------------
// Kernel 1: conv1d(k=3,pad=1) as GEMM.  out[b,t,h] = sum_{d,kt} x[b,t+kt-1,d]*w[h,d,kt]
// One wave -> 16(t) x 64(h) strip: 4 accumulators, A fragment reused 4x.
// Boundary tiles (t0==0, t0==T-16) take a value-masked path; interior tiles
// take the maskless path (fmul x,1.0 folds), selected by a block-uniform branch.
// Writes h fp32 [B,T,H] (== v) and vT bf16 [B,H,T] for the feat GEMM.
// ---------------------------------------------------------------------------
template <bool MASKED>
__device__ __forceinline__ void conv_body(
    const float* __restrict__ x, const __bf16* __restrict__ wpk,
    v8f acc[4], int b, int t0, int n0, int m, int hi, int T, int D, int H) {
  for (int kt = 0; kt < 3; ++kt) {
    const int t = t0 + m + kt - 1;
    const bool okT = (t >= 0) && (t < T);
    const int tc = MASKED ? (okT ? t : 0) : t;          // safe clamped address
    const float msk = MASKED ? (okT ? 1.0f : 0.0f) : 1.0f;
    const float*  xrow = x + ((size_t)(b * T + tc)) * D + hi * 16;
    const __bf16* wrow = wpk + ((size_t)kt * H + n0 + m) * D + hi * 16;
    for (int d0 = 0; d0 < D; d0 += 32) {
      __builtin_prefetch(xrow + d0 + 512, 0, 1);        // global_prefetch_b8
      const float4 l0 = *(const float4*)(xrow + d0);
      const float4 l1 = *(const float4*)(xrow + d0 + 4);
      const float4 h0 = *(const float4*)(xrow + d0 + 8);
      const float4 h1 = *(const float4*)(xrow + d0 + 12);
      v16bf a;
      a[0]  = (__bf16)(l0.x * msk); a[1]  = (__bf16)(l0.y * msk);
      a[2]  = (__bf16)(l0.z * msk); a[3]  = (__bf16)(l0.w * msk);
      a[4]  = (__bf16)(l1.x * msk); a[5]  = (__bf16)(l1.y * msk);
      a[6]  = (__bf16)(l1.z * msk); a[7]  = (__bf16)(l1.w * msk);
      a[8]  = (__bf16)(h0.x * msk); a[9]  = (__bf16)(h0.y * msk);
      a[10] = (__bf16)(h0.z * msk); a[11] = (__bf16)(h0.w * msk);
      a[12] = (__bf16)(h1.x * msk); a[13] = (__bf16)(h1.y * msk);
      a[14] = (__bf16)(h1.z * msk); a[15] = (__bf16)(h1.w * msk);
#pragma unroll
      for (int q = 0; q < 4; ++q) {                     // A reused across 4 B tiles
        const v16bf bm = *(const v16bf*)(wrow + (size_t)q * 16 * D + d0);
        acc[q] = __builtin_amdgcn_wmma_f32_16x16x32_bf16(
            false, a, false, bm, (short)0, acc[q], false, false);
      }
    }
  }
}

__global__ __launch_bounds__(128) void conv_gemm(
    const float* __restrict__ x, const __bf16* __restrict__ wpk,
    float* __restrict__ hout, __bf16* __restrict__ vT,
    int B, int T, int D, int H) {
  const int wave = threadIdx.x >> 5, lane = threadIdx.x & 31;
  const int m = lane & 15, hi = lane >> 4;
  const int tilesT = T >> 4;
  const int b  = blockIdx.x / tilesT;
  const int t0 = (blockIdx.x % tilesT) << 4;
  const int n0 = ((blockIdx.y << 2) + wave) << 6;       // 64-wide strip per wave

  v8f acc[4] = {};
  if (t0 != 0 && t0 != T - 16)
    conv_body<false>(x, wpk, acc, b, t0, n0, m, hi, T, D, H);
  else
    conv_body<true>(x, wpk, acc, b, t0, n0, m, hi, T, D, H);

#pragma unroll
  for (int q = 0; q < 4; ++q) {
    const int n = n0 + q * 16 + m;
#pragma unroll
    for (int r = 0; r < 8; ++r) {
      const int t = t0 + hi * 8 + r;
      hout[((size_t)(b * T + t)) * H + n] = acc[q][r];
    }
    v8bf pk;
#pragma unroll
    for (int r = 0; r < 8; ++r) pk[r] = (__bf16)acc[q][r];
    *(v8bf*)(vT + ((size_t)b * H + n) * T + t0 + hi * 8) = pk;   // 16B store
  }
}

// ---------------------------------------------------------------------------
// Kernel 2: channel-L2-normalize each (b,t) row of h -> xn bf16 [B,T,H]
// ---------------------------------------------------------------------------
__global__ void rownorm_xn(const float* __restrict__ h, __bf16* __restrict__ xn,
                           int H) {
  const int row = blockIdx.x, lane = threadIdx.x;
  const float* r = h + (size_t)row * H;
  float ss = 0.f;
  for (int i = lane; i < H; i += 32) { float v = r[i]; ss += v * v; }
  ss = wave_sum(ss);
  const float inv = 1.0f / fmaxf(sqrtf(ss), EPSF);
  __bf16* o = xn + (size_t)row * H;
  for (int i = lane; i < H; i += 32) o[i] = (__bf16)(r[i] * inv);
}

// ---------------------------------------------------------------------------
// Kernel 3: sim[b,t,s] = sum_c xn[b,t,c]*xn[b,s,c]   (Xn * Xn^T, WMMA)
// One wave -> 16(t) x 64(s) strip, A fragment reused 4x.
// ---------------------------------------------------------------------------
__global__ __launch_bounds__(128) void sim_gemm(
    const __bf16* __restrict__ xn, float* __restrict__ sim, int T, int H) {
  const int wave = threadIdx.x >> 5, lane = threadIdx.x & 31;
  const int m = lane & 15, hi = lane >> 4;
  const int b  = blockIdx.z;
  const int t0 = blockIdx.x << 4;
  const int s0 = ((blockIdx.y << 2) + wave) << 6;
  const __bf16* base = xn + (size_t)b * T * H;
  const __bf16* arow = base + (size_t)(t0 + m) * H + hi * 16;
  const __bf16* brow = base + (size_t)(s0 + m) * H + hi * 16;

  v8f acc[4] = {};
  for (int c0 = 0; c0 < H; c0 += 32) {
    const v8bf a0 = *(const v8bf*)(arow + c0);
    const v8bf a1 = *(const v8bf*)(arow + c0 + 8);
    const v16bf a = cat16(a0, a1);
#pragma unroll
    for (int q = 0; q < 4; ++q) {
      const v16bf bm = *(const v16bf*)(brow + (size_t)q * 16 * H + c0);
      acc[q] = __builtin_amdgcn_wmma_f32_16x16x32_bf16(
          false, a, false, bm, (short)0, acc[q], false, false);
    }
  }
#pragma unroll
  for (int q = 0; q < 4; ++q)
#pragma unroll
    for (int r = 0; r < 8; ++r) {
      const size_t off =
          ((size_t)b * T + (t0 + hi * 8 + r)) * (size_t)T + s0 + q * 16 + m;
      sim[off] = acc[q][r];
    }
}

// ---------------------------------------------------------------------------
// Kernel 4: per row of sim: exact k-th-largest via 32-step radix search on
// sortable keys, threshold (>=), row-normalize, emit g bf16 [B,T,T].
// One wave per row; each lane holds 32 elements (t = lane + 32*j).
// ---------------------------------------------------------------------------
__global__ void build_g(const float* __restrict__ sim, __bf16* __restrict__ g,
                        int T, int k) {
  const int row = blockIdx.x, lane = threadIdx.x;
  const float* r = sim + (size_t)row * T;
  float v[32]; unsigned key[32];
#pragma unroll
  for (int j = 0; j < 32; ++j) { v[j] = r[lane + 32 * j]; key[j] = sortkey(v[j]); }

  unsigned res = 0;
  for (int bit = 31; bit >= 0; --bit) {
    const unsigned cand = res | (1u << bit);
    int cnt = 0;
#pragma unroll
    for (int j = 0; j < 32; ++j) cnt += (key[j] >= cand) ? 1 : 0;
    cnt = wave_sum_i(cnt);
    if (cnt >= k) res = cand;      // uniform across wave
  }
  float s = 0.f;
#pragma unroll
  for (int j = 0; j < 32; ++j) if (key[j] >= res) s += v[j];
  s = wave_sum(s);
  const float rs = (s == 0.f) ? 1.f : s;
  __bf16* go = g + (size_t)row * T;
#pragma unroll
  for (int j = 0; j < 32; ++j) {
    const float gv = (key[j] >= res) ? (v[j] / rs) : 0.f;
    go[lane + 32 * j] = (__bf16)gv;
  }
}

// ---------------------------------------------------------------------------
// Kernel 5: feat = g @ v + v  (WMMA; B from vT so loads are contiguous;
// one wave -> 16(t) x 64(h) strip, A fragment reused 4x)
// ---------------------------------------------------------------------------
__global__ __launch_bounds__(128) void feat_gemm(
    const __bf16* __restrict__ g, const __bf16* __restrict__ vT,
    const float* __restrict__ hsrc, float* __restrict__ feat, int T, int H) {
  const int wave = threadIdx.x >> 5, lane = threadIdx.x & 31;
  const int m = lane & 15, hi = lane >> 4;
  const int tilesT = T >> 4;
  const int b  = blockIdx.x / tilesT;
  const int t0 = (blockIdx.x % tilesT) << 4;
  const int n0 = ((blockIdx.y << 2) + wave) << 6;
  const __bf16* arow = g + ((size_t)b * T + t0 + m) * T + hi * 16;
  const __bf16* brow = vT + ((size_t)b * H + n0 + m) * T + hi * 16;

  v8f acc[4] = {};
  for (int c0 = 0; c0 < T; c0 += 32) {
    const v8bf a0 = *(const v8bf*)(arow + c0);
    const v8bf a1 = *(const v8bf*)(arow + c0 + 8);
    const v16bf a = cat16(a0, a1);
#pragma unroll
    for (int q = 0; q < 4; ++q) {
      const v16bf bm = *(const v16bf*)(brow + (size_t)q * 16 * T + c0);
      acc[q] = __builtin_amdgcn_wmma_f32_16x16x32_bf16(
          false, a, false, bm, (short)0, acc[q], false, false);
    }
  }
#pragma unroll
  for (int q = 0; q < 4; ++q)
#pragma unroll
    for (int r = 0; r < 8; ++r) {
      const int t = t0 + hi * 8 + r;
      const size_t off = ((size_t)(b * T + t)) * H + n0 + q * 16 + m;
      feat[off] = acc[q][r] + hsrc[off];
    }
}

// ---------------------------------------------------------------------------
// Kernel 6: pn[h,c] = proxies[h,c] / ||proxies[:,c]||
// ---------------------------------------------------------------------------
__global__ void pn_kernel(const float* __restrict__ p, float* __restrict__ pn,
                          int H, int C) {
  const int c = blockIdx.x, lane = threadIdx.x;
  float ss = 0.f;
  for (int hh = lane; hh < H; hh += 32) { float v = p[hh * C + c]; ss += v * v; }
  ss = wave_sum(ss);
  const float inv = 1.0f / fmaxf(sqrtf(ss), EPSF);
  for (int hh = lane; hh < H; hh += 32) pn[hh * C + c] = p[hh * C + c] * inv;
}

// ---------------------------------------------------------------------------
// Kernel 7: per (b,t): sas = ||feat row||; cas[c] = <featn,pn[:,c]>/TEMP;
// cas_score = softmax_c(cas).  Feat row cached in 32 regs/lane; the 20 dots
// run wave-parallel with shuffle reductions (pn read at stride C across lanes).
// ---------------------------------------------------------------------------
__global__ void cas_sas(const float* __restrict__ feat, const float* __restrict__ pn,
                        float* __restrict__ casraw, float* __restrict__ cas_score,
                        float* __restrict__ sas_ws, int C) {
  const int row = blockIdx.x, lane = threadIdx.x;
  const float* f = feat + (size_t)row * 1024;
  float fr[32];
#pragma unroll
  for (int i = 0; i < 32; ++i) fr[i] = f[lane + 32 * i];
  float ss = 0.f;
#pragma unroll
  for (int i = 0; i < 32; ++i) ss += fr[i] * fr[i];
  ss = wave_sum(ss);
  const float nrm = sqrtf(ss);
  if (lane == 0) sas_ws[row] = nrm;
  const float scl = (1.0f / fmaxf(nrm, EPSF)) * (1.0f / TEMP);

  float casv = -__builtin_inff();
  for (int c = 0; c < C; ++c) {
    float dot = 0.f;
#pragma unroll
    for (int i = 0; i < 32; ++i) dot += fr[i] * pn[(lane + 32 * i) * C + c];
    dot = wave_sum(dot);                    // uniform
    const float val = dot * scl;
    if (lane == c) casv = val;
  }
  if (lane < C) casraw[(size_t)row * C + lane] = casv;
  const float mx = wave_max(casv);
  const float ex = (lane < C) ? __expf(casv - mx) : 0.f;
  const float sm = wave_sum(ex);
  if (lane < C) cas_score[(size_t)row * C + lane] = ex / sm;
}

// ---------------------------------------------------------------------------
// Kernel 8: per-batch min/max of sas over T
// ---------------------------------------------------------------------------
__global__ void sas_minmax(const float* __restrict__ sas, float* __restrict__ mm,
                           int T) {
  const int b = blockIdx.x, lane = threadIdx.x;
  float mn = __builtin_inff(), mx = -__builtin_inff();
  for (int t = lane; t < T; t += 32) {
    const float v = sas[(size_t)b * T + t];
    mn = fminf(mn, v); mx = fmaxf(mx, v);
  }
  mn = wave_min(mn); mx = wave_max(mx);
  if (lane == 0) { mm[b * 2] = mn; mm[b * 2 + 1] = mx; }
}

// ---------------------------------------------------------------------------
// Kernel 9: elementwise seg_score + sas_score
// ---------------------------------------------------------------------------
__global__ void seg_kernel(const float* __restrict__ cas_score,
                           const float* __restrict__ sas, const float* __restrict__ mm,
                           float* __restrict__ seg, float* __restrict__ sas_score,
                           int B, int T, int C) {
  const int idx = blockIdx.x * blockDim.x + threadIdx.x;
  if (idx >= B * T * C) return;
  const int c = idx % C, bt = idx / C, b = bt / T;
  const float mn = mm[b * 2], mx = mm[b * 2 + 1];
  const float sss = (sas[bt] - mn) / ((mx == 0.f) ? 1.f : mx);
  if (c == 0) sas_score[bt] = sss;
  seg[idx] = 0.5f * (cas_score[idx] + sss);
}

// ---------------------------------------------------------------------------
// Kernel 10: per (b,c): top-k (and bottom-k) time indices of seg (jax top_k
// tie-break: lower index wins). One wave per (b,c); iterative arg-max with a
// 32-bit used-mask per lane (t = lane + 32*j).
// ---------------------------------------------------------------------------
__global__ void topk_sel(const float* __restrict__ seg, const float* __restrict__ casraw,
                         int* __restrict__ act_index, float* __restrict__ act_mean,
                         float* __restrict__ bkg_mean, int T, int C, int k) {
  const int b = blockIdx.x / C, c = blockIdx.x % C;
  const int lane = threadIdx.x;
  float v[32];
#pragma unroll
  for (int j = 0; j < 32; ++j) {
    const int t = lane + 32 * j;
    v[j] = seg[((size_t)b * T + t) * C + c];
  }
  // ---- act: largest k ----
  unsigned mask = 0xFFFFFFFFu; float asum = 0.f;
  for (int it = 0; it < k; ++it) {
    float best = -__builtin_inff(); int bidx = T;
#pragma unroll
    for (int j = 0; j < 32; ++j)
      if (mask & (1u << j)) {
        const int t = lane + 32 * j;
        if (v[j] > best) { best = v[j]; bidx = t; }
      }
#pragma unroll
    for (int m = 16; m >= 1; m >>= 1) {
      const float ov = __shfl_xor(best, m, 32);
      const int   oi = __shfl_xor(bidx, m, 32);
      if (ov > best || (ov == best && oi < bidx)) { best = ov; bidx = oi; }
    }
    if ((bidx & 31) == lane) mask &= ~(1u << (bidx >> 5));
    if (lane == 0) {
      act_index[((size_t)b * k + it) * C + c] = bidx;
      asum += casraw[((size_t)b * T + bidx) * C + c];
    }
  }
  if (lane == 0) act_mean[b * C + c] = asum / (float)k;
  // ---- bkg: smallest k (top_k of -seg) ----
  mask = 0xFFFFFFFFu; float bsum = 0.f;
  for (int it = 0; it < k; ++it) {
    float best = -__builtin_inff(); int bidx = T;
#pragma unroll
    for (int j = 0; j < 32; ++j)
      if (mask & (1u << j)) {
        const int t = lane + 32 * j;
        if (-v[j] > best) { best = -v[j]; bidx = t; }
      }
#pragma unroll
    for (int m = 16; m >= 1; m >>= 1) {
      const float ov = __shfl_xor(best, m, 32);
      const int   oi = __shfl_xor(bidx, m, 32);
      if (ov > best || (ov == best && oi < bidx)) { best = ov; bidx = oi; }
    }
    if ((bidx & 31) == lane) mask &= ~(1u << (bidx >> 5));
    if (lane == 0) bsum += casraw[((size_t)b * T + bidx) * C + c];
  }
  if (lane == 0) bkg_mean[b * C + c] = bsum / (float)k;
}

// ---------------------------------------------------------------------------
// Kernel 11: per-batch softmax over C of mean cas -> act/bkg score
// ---------------------------------------------------------------------------
__global__ void final_softmax(const float* __restrict__ mean, float* __restrict__ score,
                              int C) {
  const int b = blockIdx.x, lane = threadIdx.x;
  const float v = (lane < C) ? mean[b * C + lane] : -__builtin_inff();
  const float mx = wave_max(v);
  const float e = (lane < C) ? __expf(v - mx) : 0.f;
  const float s = wave_sum(e);
  if (lane < C) score[b * C + lane] = e / s;
}

// ---------------------------------------------------------------------------
extern "C" void kernel_launch(void* const* d_in, const int* in_sizes, int n_in,
                              void* d_out, int out_size, void* d_ws, size_t ws_size,
                              hipStream_t stream) {
  constexpr int B = 16, T = 1024, D = 2048, H = 1024, C = 20, K = T / 8;

  const float* x       = (const float*)d_in[0];
  const float* conv_w  = (const float*)d_in[1];
  const float* proxies = (const float*)d_in[2];

  // output layout (reference tuple order, flat)
  float* out       = (float*)d_out;
  float* act_score = out;                                    // B*C
  float* bkg_score = out + (size_t)B * C;                    // B*C
  float* cas_score = bkg_score + (size_t)B * C;              // B*T*C
  float* sas_score = cas_score + (size_t)B * T * C;          // B*T
  float* seg_score = sas_score + (size_t)B * T;              // B*T*C
  int*   act_index = (int*)(seg_score + (size_t)B * T * C);  // B*K*C (int32)
  float* sim       = (float*)(act_index + (size_t)B * K * C);// B*T*T

  // workspace carve-up
  char* w = (char*)d_ws;
  float*  hbuf   = (float*)w;  w += (size_t)B * T * H * sizeof(float);
  __bf16* vT     = (__bf16*)w; w += (size_t)B * H * T * sizeof(__bf16);
  __bf16* xnbf   = (__bf16*)w; w += (size_t)B * T * H * sizeof(__bf16);
  __bf16* gbf    = (__bf16*)w; w += (size_t)B * T * T * sizeof(__bf16);
  float*  feat   = (float*)w;  w += (size_t)B * T * H * sizeof(float);
  __bf16* wpk    = (__bf16*)w; w += (size_t)3 * H * D * sizeof(__bf16);
  float*  casraw = (float*)w;  w += (size_t)B * T * C * sizeof(float);
  float*  sasws  = (float*)w;  w += (size_t)B * T * sizeof(float);
  float*  pn     = (float*)w;  w += (size_t)H * C * sizeof(float);
  float*  mm     = (float*)w;  w += (size_t)B * 2 * sizeof(float);
  float*  actm   = (float*)w;  w += (size_t)B * C * sizeof(float);
  float*  bkgm   = (float*)w;  w += (size_t)B * C * sizeof(float);

  // 0. repack conv weights -> bf16 [3,H,D]
  repack_w<<<2048, 256, 0, stream>>>(conv_w, wpk, H, D);
  // 1. conv as WMMA GEMM (writes h fp32 [B,T,H] and vT bf16 [B,H,T])
  conv_gemm<<<dim3(B * (T / 16), H / 256), 128, 0, stream>>>(x, wpk, hbuf, vT, B, T, D, H);
  // 2. channel L2 normalize
  rownorm_xn<<<B * T, 32, 0, stream>>>(hbuf, xnbf, H);
  // 3. similarity graph (writes directly into d_out sim slice)
  sim_gemm<<<dim3(T / 16, T / 256, B), 128, 0, stream>>>(xnbf, sim, T, H);
  // 4. k-th-largest threshold + row normalize -> g
  build_g<<<B * T, 32, 0, stream>>>(sim, gbf, T, K);
  // 5. feat = g @ v + v
  feat_gemm<<<dim3(B * (T / 16), H / 256), 128, 0, stream>>>(gbf, vT, hbuf, feat, T, H);
  // 6. proxy normalize
  pn_kernel<<<C, 32, 0, stream>>>(proxies, pn, H, C);
  // 7. cas / cas_score / sas
  cas_sas<<<B * T, 32, 0, stream>>>(feat, pn, casraw, cas_score, sasws, C);
  // 8. per-batch min/max of sas
  sas_minmax<<<B, 32, 0, stream>>>(sasws, mm, T);
  // 9. seg_score + sas_score
  seg_kernel<<<(B * T * C + 255) / 256, 256, 0, stream>>>(cas_score, sasws, mm,
                                                          seg_score, sas_score, B, T, C);
  // 10. per-(b,c) top-k act/bkg selection
  topk_sel<<<B * C, 32, 0, stream>>>(seg_score, casraw, act_index, actm, bkgm, T, C, K);
  // 11. final softmaxes
  final_softmax<<<B, 32, 0, stream>>>(actm, act_score, C);
  final_softmax<<<B, 32, 0, stream>>>(bkgm, bkg_score, C);
}
